// Seq2SeqWithAttention_64570538328568
// MI455X (gfx1250) — matrix operand order repaired and checked by reading, not compiled, not run on hardware
//
#include <hip/hip_runtime.h>
#include <math.h>

#define B_  32
#define S_  128
#define T_  64
#define V_  32000
#define E_  256
#define H_  512
#define H2_ 1024
#define G1_ 2048   // 4*H
#define G2_ 4096   // 4*H2

typedef __bf16 bf16_t;
typedef __attribute__((ext_vector_type(16))) __bf16 v16bf;
typedef __attribute__((ext_vector_type(8)))  __bf16 v8bf;
typedef __attribute__((ext_vector_type(8)))  float  v8f;

__device__ __forceinline__ float sigf(float x) { return 1.f / (1.f + __expf(-x)); }

// ---------------------------------------------------------------------------
// Generic bf16 WMMA GEMM:  D[M,N] (f32) (+)= A[M,K] @ W[N,K]^T (+ bias[N])
// A row-major bf16 (lda elems), W row-major bf16 (ldw elems), D row-major f32.
// Block = 256 threads = 8 waves; block tile 32x256.
// Each wave computes a 16x64 strip as 4 WMMA accumulators sharing one A
// fragment per K-step: 32B(A)+128B(B) per 4 WMMAs. Inner loop is branch-free:
// weight prefetches are speculative (OOB prefetch is dropped in HW), so no
// tail guard is needed and the loop stays a single basic block.
// Requires M % 16 == 0, N % 256 == 0, K % 32 == 0 (true for all uses here).
// ---------------------------------------------------------------------------
__global__ __launch_bounds__(256)
void wmma_gemm_kernel(const bf16_t* __restrict__ A, int lda,
                      const bf16_t* __restrict__ W, int ldw,
                      float* __restrict__ D, long long ldd,
                      const float* __restrict__ bias,
                      int M, int N, int K, int accumulate)
{
    const int tid  = threadIdx.x;
    const int wave = tid >> 5;
    const int lane = tid & 31;
    const int half = lane >> 4;
    const int l16  = lane & 15;

    const int m0 = blockIdx.y * 32 + (wave & 1) * 16;
    const int n0 = blockIdx.x * 256 + (wave >> 1) * 64;
    if (m0 >= M || n0 >= N) return;      // wave-uniform

    const bf16_t* Arow = A + (size_t)(m0 + l16) * (size_t)lda;
    const bf16_t* W0 = W + (size_t)(n0 +  0 + l16) * (size_t)ldw;
    const bf16_t* W1 = W + (size_t)(n0 + 16 + l16) * (size_t)ldw;
    const bf16_t* W2 = W + (size_t)(n0 + 32 + l16) * (size_t)ldw;
    const bf16_t* W3 = W + (size_t)(n0 + 48 + l16) * (size_t)ldw;

    v8f acc0 = {}, acc1 = {}, acc2 = {}, acc3 = {};
    for (int k0 = 0; k0 < K; k0 += 32) {
        // A fragment (16x32): row m = lane%16, K = half*16+v (contig, matches B)
        union { v16bf v; v8bf h[2]; } a;
        a.h[0] = *(const v8bf*)(Arow + k0 + half * 16);
        a.h[1] = *(const v8bf*)(Arow + k0 + half * 16 + 8);
        // B fragments (32x16): lane n = lane%16, K = half*16+v (contig in row)
        const v16bf b0 = *(const v16bf*)(W0 + k0 + half * 16);
        const v16bf b1 = *(const v16bf*)(W1 + k0 + half * 16);
        const v16bf b2 = *(const v16bf*)(W2 + k0 + half * 16);
        const v16bf b3 = *(const v16bf*)(W3 + k0 + half * 16);
        // speculative prefetch two K-chunks ahead (branch-free, OOB-safe)
        __builtin_prefetch(W0 + k0 + 64, 0, 1);
        __builtin_prefetch(W1 + k0 + 64, 0, 1);
        __builtin_prefetch(W2 + k0 + 64, 0, 1);
        __builtin_prefetch(W3 + k0 + 64, 0, 1);
        acc0 = __builtin_amdgcn_wmma_f32_16x16x32_bf16(false, a.v, false, b0,
                                                       (short)0, acc0, false, false);
        acc1 = __builtin_amdgcn_wmma_f32_16x16x32_bf16(false, a.v, false, b1,
                                                       (short)0, acc1, false, false);
        acc2 = __builtin_amdgcn_wmma_f32_16x16x32_bf16(false, a.v, false, b2,
                                                       (short)0, acc2, false, false);
        acc3 = __builtin_amdgcn_wmma_f32_16x16x32_bf16(false, a.v, false, b3,
                                                       (short)0, acc3, false, false);
    }

    const v8f accs[4] = { acc0, acc1, acc2, acc3 };
#pragma unroll
    for (int j = 0; j < 4; ++j) {
        const int n = n0 + j * 16 + l16;
        const float badd = bias ? bias[n] : 0.f;
#pragma unroll
        for (int r = 0; r < 8; ++r) {
            const int m = m0 + r + half * 8;
            const size_t idx = (size_t)m * (size_t)ldd + (size_t)n;
            float val = accs[j][r] + badd;
            if (accumulate) val += D[idx];
            D[idx] = val;
        }
    }
}

// ---------------------------------------------------------------------------
__global__ void f32_to_bf16_kernel(const float* __restrict__ s,
                                   bf16_t* __restrict__ d, int n)
{
    int i = blockIdx.x * 256 + threadIdx.x;
    if (i < n) d[i] = (bf16_t)s[i];
}

// gather emb_enc[src[b,s]] -> Xe[s,b,:] bf16 ; one block per (s*B+b), 256 thr
__global__ void embed_enc_kernel(const int* __restrict__ src,
                                 const float* __restrict__ emb,
                                 bf16_t* __restrict__ Xe)
{
    const int sb = blockIdx.x;           // s*B + b
    const int b = sb % B_, s = sb / B_;
    const int tok = src[b * S_ + s];
    Xe[(size_t)sb * E_ + threadIdx.x] =
        (bf16_t)emb[(size_t)tok * E_ + threadIdx.x];
}

// fused LSTM cell elementwise update. g = (B,4*Hd) preactivation (xW + hW).
__global__ void lstm_cell_kernel(const float* __restrict__ g,
                                 const float* __restrict__ bih,
                                 const float* __restrict__ bhh,
                                 float* __restrict__ c,
                                 bf16_t* __restrict__ h_bf, int h_stride,
                                 float* __restrict__ h_f32,
                                 bf16_t* __restrict__ hs_out, int hs_stride,
                                 int Hd, int first)
{
    const int i = blockIdx.x * 256 + threadIdx.x;
    if (i >= B_ * Hd) return;
    const int b = i / Hd, j = i % Hd;
    const float* gr = g + (size_t)b * 4 * Hd;
    const float gi = gr[j]          + bih[j]          + bhh[j];
    const float gf = gr[Hd + j]     + bih[Hd + j]     + bhh[Hd + j];
    const float gg = gr[2 * Hd + j] + bih[2 * Hd + j] + bhh[2 * Hd + j];
    const float go = gr[3 * Hd + j] + bih[3 * Hd + j] + bhh[3 * Hd + j];
    const float cprev = first ? 0.f : c[i];
    const float cn = sigf(gf) * cprev + sigf(gi) * tanhf(gg);
    const float hn = sigf(go) * tanhf(cn);
    c[i] = cn;
    h_bf[(size_t)b * h_stride + j] = (bf16_t)hn;
    if (h_f32)  h_f32[i] = hn;
    if (hs_out) hs_out[(size_t)b * hs_stride + j] = (bf16_t)hn;
}

// build decoder init state: h[l] = concat(fwd_l, bwd_l), same for c
__global__ void dec_init_kernel(const float* __restrict__ hf0, const float* __restrict__ hb0,
                                const float* __restrict__ hf1, const float* __restrict__ hb1,
                                const float* __restrict__ cf0, const float* __restrict__ cb0,
                                const float* __restrict__ cf1, const float* __restrict__ cb1,
                                bf16_t* __restrict__ hdec, float* __restrict__ cdec)
{
    const int i = blockIdx.x * 256 + threadIdx.x;
    if (i >= 2 * B_ * H2_) return;
    const int l = i / (B_ * H2_);
    const int rem = i % (B_ * H2_);
    const int b = rem / H2_, j = rem % H2_;
    const int jj = j & (H_ - 1);
    const float* hsrc = (l == 0) ? (j < H_ ? hf0 : hb0) : (j < H_ ? hf1 : hb1);
    const float* csrc = (l == 0) ? (j < H_ ? cf0 : cb0) : (j < H_ ? cf1 : cb1);
    hdec[i] = (bf16_t)hsrc[b * H_ + jj];
    cdec[i] = csrc[b * H_ + jj];
}

// fused attention + decoder-input build. one block per batch elem, 128 threads.
// scores[s] = v . tanh(enc_proj[s,b,:] + h_proj[b,:])   (ba folded into enc_proj)
// softmax over S, ctx = sum_s w[s]*enc_out[s,b,:]; xin = concat(emb_dec[tok], ctx)
__global__ __launch_bounds__(128)
void attend_kernel(const float* __restrict__ enc_proj,   // (S,B,H)
                   const float* __restrict__ h_proj,     // (B,H)
                   const float* __restrict__ vvec,       // (H)
                   const bf16_t* __restrict__ enc_out,   // (S,B,2H) bf16
                   const float* __restrict__ emb_dec,    // (V,E)
                   const int* __restrict__ tgt, int t,
                   bf16_t* __restrict__ xin)             // (B, E+2H) bf16
{
    __shared__ float sh_h[H_];
    __shared__ float sh_v[H_];
    __shared__ float sc[S_];
    __shared__ float red[S_];
    const int b = blockIdx.x;
    const int tx = threadIdx.x;

    for (int j = tx; j < H_; j += 128) { sh_h[j] = h_proj[b * H_ + j]; sh_v[j] = vvec[j]; }
    __syncthreads();

    {   // one score per thread (S_ == blockDim)
        const float* ep = enc_proj + ((size_t)tx * B_ + b) * H_;
        float acc = 0.f;
        for (int j = 0; j < H_; ++j) acc += sh_v[j] * tanhf(ep[j] + sh_h[j]);
        sc[tx] = acc;
    }
    __syncthreads();

    red[tx] = sc[tx];
    __syncthreads();
    for (int off = 64; off; off >>= 1) {
        if (tx < off) red[tx] = fmaxf(red[tx], red[tx + off]);
        __syncthreads();
    }
    const float mx = red[0];
    __syncthreads();
    const float e = __expf(sc[tx] - mx);
    red[tx] = e;
    __syncthreads();
    for (int off = 64; off; off >>= 1) {
        if (tx < off) red[tx] += red[tx + off];
        __syncthreads();
    }
    const float inv = 1.f / red[0];
    __syncthreads();
    sc[tx] = e * inv;                    // softmax weights
    __syncthreads();

    // context -> xin[:, E: E+2H]
    for (int d = tx; d < H2_; d += 128) {
        float acc = 0.f;
        for (int s = 0; s < S_; ++s)
            acc += sc[s] * (float)enc_out[((size_t)s * B_ + b) * H2_ + d];
        xin[(size_t)b * (E_ + H2_) + E_ + d] = (bf16_t)acc;
    }
    // embedding -> xin[:, 0:E]
    const int tok = tgt[b * T_ + t];
    for (int e2 = tx; e2 < E_; e2 += 128)
        xin[(size_t)b * (E_ + H2_) + e2] = (bf16_t)emb_dec[(size_t)tok * E_ + e2];
}

__global__ void zero_t0_kernel(float* __restrict__ out)
{
    const size_t i = (size_t)blockIdx.x * 256 + threadIdx.x;
    if (i < (size_t)B_ * V_) {
        const size_t b = i / V_, n = i % V_;
        out[b * (size_t)T_ * V_ + n] = 0.f;
    }
}

// ---------------------------------------------------------------------------
extern "C" void kernel_launch(void* const* d_in, const int* in_sizes, int n_in,
                              void* d_out, int out_size, void* d_ws, size_t ws_size,
                              hipStream_t stream)
{
    (void)in_sizes; (void)n_in; (void)out_size; (void)ws_size;

    const int*   src     = (const int*)d_in[0];
    const int*   tgt     = (const int*)d_in[1];
    const float* emb_enc = (const float*)d_in[2];
    const float* emb_dec = (const float*)d_in[3];
    // encoder LSTMs: 0=l0 fwd, 1=l0 bwd, 2=l1 fwd, 3=l1 bwd  (base idx 4,8,12,16)
    const float *eWih[4], *eWhh[4], *ebih[4], *ebhh[4];
    for (int u = 0; u < 4; ++u) {
        eWih[u] = (const float*)d_in[4 + 4 * u + 0];
        eWhh[u] = (const float*)d_in[4 + 4 * u + 1];
        ebih[u] = (const float*)d_in[4 + 4 * u + 2];
        ebhh[u] = (const float*)d_in[4 + 4 * u + 3];
    }
    const float* dWih0 = (const float*)d_in[20];
    const float* dWhh0 = (const float*)d_in[21];
    const float* dbih0 = (const float*)d_in[22];
    const float* dbhh0 = (const float*)d_in[23];
    const float* dWih1 = (const float*)d_in[24];
    const float* dWhh1 = (const float*)d_in[25];
    const float* dbih1 = (const float*)d_in[26];
    const float* dbhh1 = (const float*)d_in[27];
    const float* Wa    = (const float*)d_in[28];
    const float* ba    = (const float*)d_in[29];
    const float* vvec  = (const float*)d_in[30];
    const float* Wout  = (const float*)d_in[31];
    const float* bout  = (const float*)d_in[32];
    float* out = (float*)d_out;

    // ---- workspace bump allocator (256B aligned) ----
    char* ws = (char*)d_ws;
    size_t off = 0;
    auto alloc = [&](size_t bytes) -> void* {
        void* p = ws + off;
        off = (off + bytes + 255) & ~(size_t)255;
        return p;
    };
    const int eWihK[4] = { E_, E_, H2_, H2_ };
    bf16_t* wWih[4]; bf16_t* wWhh[4];
    for (int u = 0; u < 4; ++u) {
        wWih[u] = (bf16_t*)alloc((size_t)G1_ * eWihK[u] * 2);
        wWhh[u] = (bf16_t*)alloc((size_t)G1_ * H_ * 2);
    }
    bf16_t* wDWih0 = (bf16_t*)alloc((size_t)G2_ * (E_ + H2_) * 2);
    bf16_t* wDWhh0 = (bf16_t*)alloc((size_t)G2_ * H2_ * 2);
    bf16_t* wDWih1 = (bf16_t*)alloc((size_t)G2_ * H2_ * 2);
    bf16_t* wDWhh1 = (bf16_t*)alloc((size_t)G2_ * H2_ * 2);
    bf16_t* wWa    = (bf16_t*)alloc((size_t)H_ * H2_ * 2);
    bf16_t* wWout  = (bf16_t*)alloc((size_t)V_ * H2_ * 2);   // 64 MB -> L2-resident
    bf16_t* Xe     = (bf16_t*)alloc((size_t)S_ * B_ * E_ * 2);
    float*  XWf    = (float*) alloc((size_t)S_ * B_ * G1_ * 4);
    float*  XWb    = (float*) alloc((size_t)S_ * B_ * G1_ * 4);
    bf16_t* X1     = (bf16_t*)alloc((size_t)S_ * B_ * H2_ * 2);
    bf16_t* EncOut = (bf16_t*)alloc((size_t)S_ * B_ * H2_ * 2);
    float*  EncProj= (float*) alloc((size_t)S_ * B_ * H_ * 4);
    bf16_t* h_bf[4]; float* h_f32[4]; float* c_f32[4];
    for (int u = 0; u < 4; ++u) {
        h_bf[u]  = (bf16_t*)alloc((size_t)B_ * H_ * 2);
        h_f32[u] = (float*) alloc((size_t)B_ * H_ * 4);
        c_f32[u] = (float*) alloc((size_t)B_ * H_ * 4);
    }
    bf16_t* hdec  = (bf16_t*)alloc((size_t)2 * B_ * H2_ * 2);
    float*  cdec  = (float*) alloc((size_t)2 * B_ * H2_ * 4);
    float*  hproj = (float*) alloc((size_t)B_ * H_ * 4);
    bf16_t* xin   = (bf16_t*)alloc((size_t)B_ * (E_ + H2_) * 2);
    float*  g0    = (float*) alloc((size_t)B_ * G2_ * 4);
    float*  g1    = (float*) alloc((size_t)B_ * G2_ * 4);

    auto conv = [&](const float* s, bf16_t* d, int n) {
        f32_to_bf16_kernel<<<(n + 255) / 256, 256, 0, stream>>>(s, d, n);
    };
    auto gemm = [&](const bf16_t* A, int lda, const bf16_t* W, int ldw,
                    float* D, long long ldd, const float* bias,
                    int M, int N, int K, int acc) {
        dim3 grid(N / 256, (M + 31) / 32);
        wmma_gemm_kernel<<<grid, 256, 0, stream>>>(A, lda, W, ldw, D, ldd, bias,
                                                   M, N, K, acc);
    };
    auto cell = [&](const float* g, const float* bih, const float* bhh,
                    float* c, bf16_t* hb, int hstride, float* hf,
                    bf16_t* hs, int hsstride, int Hd, int first) {
        lstm_cell_kernel<<<(B_ * Hd + 255) / 256, 256, 0, stream>>>(
            g, bih, bhh, c, hb, hstride, hf, hs, hsstride, Hd, first);
    };

    // ---- phase 0: convert weights to bf16 ----
    for (int u = 0; u < 4; ++u) {
        conv(eWih[u], wWih[u], G1_ * eWihK[u]);
        conv(eWhh[u], wWhh[u], G1_ * H_);
    }
    conv(dWih0, wDWih0, G2_ * (E_ + H2_));
    conv(dWhh0, wDWhh0, G2_ * H2_);
    conv(dWih1, wDWih1, G2_ * H2_);
    conv(dWhh1, wDWhh1, G2_ * H2_);
    conv(Wa,    wWa,    H_ * H2_);
    conv(Wout,  wWout,  V_ * H2_);

    // ---- phase 1: encoder embedding -> Xe (S,B,E) bf16 ----
    embed_enc_kernel<<<S_ * B_, E_, 0, stream>>>(src, emb_enc, Xe);

    // ---- phase 2: encoder layer 0 (bi-dir) ----
    gemm(Xe, E_, wWih[0], E_, XWf, G1_, nullptr, S_ * B_, G1_, E_, 0);
    gemm(Xe, E_, wWih[1], E_, XWb, G1_, nullptr, S_ * B_, G1_, E_, 0);
    for (int t = 0; t < S_; ++t) {
        const int tf = t, tb = S_ - 1 - t;
        float* gF = XWf + (size_t)tf * B_ * G1_;
        float* gB = XWb + (size_t)tb * B_ * G1_;
        if (t) gemm(h_bf[0], H_, wWhh[0], H_, gF, G1_, nullptr, B_, G1_, H_, 1);
        cell(gF, ebih[0], ebhh[0], c_f32[0], h_bf[0], H_, h_f32[0],
             X1 + (size_t)tf * B_ * H2_, H2_, H_, t == 0);
        if (t) gemm(h_bf[1], H_, wWhh[1], H_, gB, G1_, nullptr, B_, G1_, H_, 1);
        cell(gB, ebih[1], ebhh[1], c_f32[1], h_bf[1], H_, h_f32[1],
             X1 + (size_t)tb * B_ * H2_ + H_, H2_, H_, t == 0);
    }

    // ---- phase 3: encoder layer 1 (bi-dir), input X1 (S,B,2H) ----
    gemm(X1, H2_, wWih[2], H2_, XWf, G1_, nullptr, S_ * B_, G1_, H2_, 0);
    gemm(X1, H2_, wWih[3], H2_, XWb, G1_, nullptr, S_ * B_, G1_, H2_, 0);
    for (int t = 0; t < S_; ++t) {
        const int tf = t, tb = S_ - 1 - t;
        float* gF = XWf + (size_t)tf * B_ * G1_;
        float* gB = XWb + (size_t)tb * B_ * G1_;
        if (t) gemm(h_bf[2], H_, wWhh[2], H_, gF, G1_, nullptr, B_, G1_, H_, 1);
        cell(gF, ebih[2], ebhh[2], c_f32[2], h_bf[2], H_, h_f32[2],
             EncOut + (size_t)tf * B_ * H2_, H2_, H_, t == 0);
        if (t) gemm(h_bf[3], H_, wWhh[3], H_, gB, G1_, nullptr, B_, G1_, H_, 1);
        cell(gB, ebih[3], ebhh[3], c_f32[3], h_bf[3], H_, h_f32[3],
             EncOut + (size_t)tb * B_ * H2_ + H_, H2_, H_, t == 0);
    }

    // ---- phase 4: hoisted attention projection  EncProj = EncOut@Wa^T + ba ----
    gemm(EncOut, H2_, wWa, H2_, EncProj, H_, ba, S_ * B_, H_, H2_, 0);

    // ---- phase 5: decoder initial state from encoder finals ----
    dec_init_kernel<<<(2 * B_ * H2_ + 255) / 256, 256, 0, stream>>>(
        h_f32[0], h_f32[1], h_f32[2], h_f32[3],
        c_f32[0], c_f32[1], c_f32[2], c_f32[3], hdec, cdec);

    // ---- phase 6: decoder loop (teacher forcing, T-1 steps) ----
    for (int t = 0; t < T_ - 1; ++t) {
        // h_proj = h[1] @ Wa^T (ba already folded into EncProj)
        gemm(hdec + (size_t)B_ * H2_, H2_, wWa, H2_, hproj, H_, nullptr,
             B_, H_, H2_, 0);
        attend_kernel<<<B_, 128, 0, stream>>>(EncProj, hproj, vvec, EncOut,
                                              emb_dec, tgt, t, xin);
        // layer 0 gates + cell
        gemm(xin,  E_ + H2_, wDWih0, E_ + H2_, g0, G2_, nullptr, B_, G2_, E_ + H2_, 0);
        gemm(hdec, H2_,      wDWhh0, H2_,      g0, G2_, nullptr, B_, G2_, H2_, 1);
        cell(g0, dbih0, dbhh0, cdec, hdec, H2_, nullptr, nullptr, 0, H2_, 0);
        // layer 1 gates + cell (uses new h0, old h1)
        gemm(hdec,                  H2_, wDWih1, H2_, g1, G2_, nullptr, B_, G2_, H2_, 0);
        gemm(hdec + (size_t)B_*H2_, H2_, wDWhh1, H2_, g1, G2_, nullptr, B_, G2_, H2_, 1);
        cell(g1, dbih1, dbhh1, cdec + (size_t)B_ * H2_,
             hdec + (size_t)B_ * H2_, H2_, nullptr, nullptr, 0, H2_, 0);
        // logits -> out[:, t+1, :]   (Wout bf16 stays hot in 192MB L2)
        gemm(hdec + (size_t)B_ * H2_, H2_, wWout, H2_,
             out + (size_t)(t + 1) * V_, (long long)T_ * V_, bout,
             B_, V_, H2_, 0);
    }

    // ---- phase 7: out[:, 0, :] = 0 ----
    zero_t0_kernel<<<((int)(((size_t)B_ * V_ + 255) / 256)), 256, 0, stream>>>(out);
}
